// MotherCubeConv_47648367182715
// MI455X (gfx1250) — matrix operand order repaired and checked by reading, not compiled, not run on hardware
//
#include <hip/hip_runtime.h>

// MotherCubeConv: out[n, o] = sum_k gathered[n, k] * W[o, k] + b[o]
//   gathered[n] = [features[n] | prev[nbr0] | prev[nbr1] | prev[nbr2] | prev[nbr3]]
// GEMM [N x 640] x [640 x 128], f16 WMMA with f32 accumulation.
// Wave tile: 4 M-tiles x 2 N-tiles (waves in 2x4 grid over the 8x8 16x16-tile space)
// -> B (weight) fragments loaded only 2x per block (L2 traffic ~0.5 TB vs 2 TB),
//    A fragments reused from LDS.

typedef __attribute__((ext_vector_type(16))) _Float16 v16h;
typedef __attribute__((ext_vector_type(8)))  _Float16 v8h;
typedef __attribute__((ext_vector_type(4)))  _Float16 v4h;
typedef __attribute__((ext_vector_type(8)))  float    v8f;
typedef __attribute__((ext_vector_type(4)))  float    v4f;

#define F        128            // F_IN == F_OUT
#define KDIM     640            // 5 * F
#define NCHUNK   5              // 128-wide K chunks: self + 4 neighbors
#define MT       128            // rows per block
#define THREADS  256            // 8 waves (wave32)
#define CSTRIDE  136            // 128 + 8 halves pad -> DS bank spread

// ---------------- kernel 0: W f32 -> f16 into workspace ----------------
__global__ void cast_w_kernel(const float* __restrict__ W,
                              _Float16* __restrict__ Wh, int n4) {
    int i = blockIdx.x * blockDim.x + threadIdx.x;
    if (i < n4) {
        v4f w = *(const v4f*)(W + (long)i * 4);
        v4h h;
        h.x = (_Float16)w.x; h.y = (_Float16)w.y;
        h.z = (_Float16)w.z; h.w = (_Float16)w.w;
        *(v4h*)(Wh + (long)i * 4) = h;
    }
}

// ---------------- main kernel ----------------
__global__ __launch_bounds__(THREADS)
void mothercube_conv_kernel(const float* __restrict__ feats,
                            const float* __restrict__ prev,
                            const int*   __restrict__ nidx,
                            const _Float16* __restrict__ Wh,   // [128 x 640] f16
                            const float* __restrict__ bias,
                            float* __restrict__ out,
                            int n_nodes)
{
    __shared__ float    s_bias[F];
    __shared__ int      s_nbr[MT * 4];
    __shared__ _Float16 sA[MT * CSTRIDE];     // one 128x128 K-chunk as f16

    const int tid  = threadIdx.x;
    const int row0 = blockIdx.x * MT;

    if (tid < F) s_bias[tid] = bias[tid];
    // preload neighbor indices for this block's rows
    for (int i = tid; i < MT * 4; i += THREADS) {
        int r = i >> 2, j = i & 3;
        int node = row0 + r;
        if (node >= n_nodes) node = n_nodes - 1;
        s_nbr[i] = nidx[node * 4 + j];
    }

    const int wave = tid >> 5;
    const int lane = tid & 31;
    const int half = lane >> 4;       // 0: lanes 0-15, 1: lanes 16-31
    const int lr   = lane & 15;
    const int mg   = wave >> 2;       // M-group (0..1): rows mg*64 .. +63
    const int ng   = wave & 3;        // N-group (0..3): cols ng*32 .. +31

    v8f acc[4][2];
    #pragma unroll
    for (int i = 0; i < 4; ++i)
        #pragma unroll
        for (int j = 0; j < 2; ++j) {
            v8f z = {0.f, 0.f, 0.f, 0.f, 0.f, 0.f, 0.f, 0.f};
            acc[i][j] = z;
        }

    for (int s = 0; s < NCHUNK; ++s) {
        // ---- stage chunk s (128 rows x 128 cols f32 -> f16 in LDS) ----
        for (int g = tid; g < MT * (F / 4); g += THREADS) {
            int r  = g >> 5;               // row in block
            int c4 = (g & 31) * 4;         // column (float4 granularity)
            int node = row0 + r;
            if (node >= n_nodes) node = n_nodes - 1;   // clamp; store predicated later
            v4f x;
            if (s == 0) {
                // features: streamed exactly once -> non-temporal, keep L2 for prev/W
                x = __builtin_nontemporal_load((const v4f*)(feats + (long)node * F + c4));
            } else {
                // prev_features: randomly gathered, L2-resident -> regular temporal
                x = *(const v4f*)(prev + (long)s_nbr[r * 4 + (s - 1)] * F + c4);
            }
            v4h h;
            h.x = (_Float16)x.x; h.y = (_Float16)x.y;
            h.z = (_Float16)x.z; h.w = (_Float16)x.w;
            *(v4h*)(sA + r * CSTRIDE + c4) = h;
        }
        __syncthreads();

        // ---- 4 K-steps of 32 over this chunk ----
        #pragma unroll
        for (int kc = 0; kc < F; kc += 32) {
            // A fragments (16-bit 16x32 layout): lanes 0-15 hold K 0..7 & 16..23,
            // lanes 16-31 hold K 8..15 & 24..31 of row M.
            union { v16h v; v8h h[2]; } af[4];
            #pragma unroll
            for (int i = 0; i < 4; ++i) {
                const _Float16* arow = sA + (mg * 64 + i * 16 + lr) * CSTRIDE + kc;
                af[i].h[0] = *(const v8h*)(arow + half * 8);
                af[i].h[1] = *(const v8h*)(arow + 16 + half * 8);
            }

            const int kg = s * F + kc;     // global K offset
            // B fragments: column N = o is W row o (K-contiguous);
            // lanes 0-15 hold K 0..7 & 16..23, lanes 16-31 hold K 8..15 & 24..31.
            union { v16h v; v8h h[2]; } bf[2];
            #pragma unroll
            for (int j = 0; j < 2; ++j) {
                const _Float16* bp =
                    Wh + (long)(ng * 32 + j * 16 + lr) * KDIM + kg + half * 8;
                bf[j].h[0] = *(const v8h*)(bp);
                bf[j].h[1] = *(const v8h*)(bp + 16);
            }

            #pragma unroll
            for (int i = 0; i < 4; ++i)
                #pragma unroll
                for (int j = 0; j < 2; ++j)
                    acc[i][j] = __builtin_amdgcn_wmma_f32_16x16x32_f16(
                        false, af[i].v, false, bf[j].v, (short)0, acc[i][j],
                        false, false);
        }
        __syncthreads();   // before next stage overwrites sA
    }

    // ---- epilogue: add bias, non-temporal f32 store (write-once stream) ----
    const int gmw = row0 + mg * 64;
    #pragma unroll
    for (int i = 0; i < 4; ++i) {
        #pragma unroll
        for (int j = 0; j < 2; ++j) {
            const int n0 = ng * 32 + j * 16;
            float bv = s_bias[n0 + lr];
            #pragma unroll
            for (int ii = 0; ii < 8; ++ii) {
                // C/D layout: VGPR ii -> row m = ii + 8*half, col = lr
                int row = gmw + i * 16 + ii + half * 8;
                if (row < n_nodes)
                    __builtin_nontemporal_store(acc[i][j][ii] + bv,
                                                &out[(long)row * F + n0 + lr]);
            }
        }
    }
}

extern "C" void kernel_launch(void* const* d_in, const int* in_sizes, int n_in,
                              void* d_out, int out_size, void* d_ws, size_t ws_size,
                              hipStream_t stream) {
    const float* feats = (const float*)d_in[0];
    const float* prev  = (const float*)d_in[1];
    const int*   nidx  = (const int*)d_in[2];
    const float* W     = (const float*)d_in[3];
    const float* bias  = (const float*)d_in[4];
    float* out = (float*)d_out;

    const int n_nodes = in_sizes[0] / F;

    // workspace: W as f16 (128*640*2 = 160 KB)
    _Float16* Wh = (_Float16*)d_ws;
    const int wn4 = (F * KDIM) / 4;            // 20480 float4 groups
    cast_w_kernel<<<(wn4 + 255) / 256, 256, 0, stream>>>(W, Wh, wn4);

    const int nblocks = (n_nodes + MT - 1) / MT;
    mothercube_conv_kernel<<<nblocks, THREADS, 0, stream>>>(
        feats, prev, nidx, Wh, bias, out, n_nodes);
}
